// MultiHeadAttention_403726926150
// MI455X (gfx1250) — matrix operand order repaired and checked by reading, not compiled, not run on hardware
//
#include <hip/hip_runtime.h>
#include <hip/hip_bf16.h>

// ---------------- problem constants ----------------
#define EMB   1024
#define SEQ   2048
#define NH    16
#define HD    64
#define BB    2
#define MTOT  (BB*SEQ)      // 4096 rows in flattened [B*N, E]

// ---------------- types ----------------
typedef __attribute__((ext_vector_type(16))) __bf16 v16bf;
typedef __attribute__((ext_vector_type(8)))  float  v8f;
typedef __attribute__((ext_vector_type(4)))  unsigned int u32x4;

union Frag16 { v16bf bf; u32x4 u[2]; };

// A-fragment (16x32 bf16, MxK): lane L holds row M=L%16,
// K elements {8h..8h+7} and {16+8h..16+8h+7}, h = L/16.
__device__ inline v16bf load_fragA(const __bf16* row, int half) {
    Frag16 f;
    f.u[0] = *reinterpret_cast<const u32x4*>(row + 8 * half);
    f.u[1] = *reinterpret_cast<const u32x4*>(row + 16 + 8 * half);
    return f.bf;
}

// B-fragment (32x16 bf16, KxN): lane L holds column N=L%16,
// K elements {8h..8h+7} and {16+8h..16+8h+7}, h = L/16.
__device__ inline v16bf load_fragB(const __bf16* colbase, int half) {
    const __bf16* p = colbase + 8 * half;
    Frag16 f;
    f.u[0] = *reinterpret_cast<const u32x4*>(p);
    f.u[1] = *reinterpret_cast<const u32x4*>(p + 16);
    return f.bf;
}

__device__ inline v8f wmma_bf16(v16bf a, v16bf b, v8f c) {
    return __builtin_amdgcn_wmma_f32_16x16x32_bf16(
        /*neg_a=*/false, a, /*neg_b=*/false, b,
        /*c_mod=*/(short)0, c, /*reuse_a=*/false, /*reuse_b=*/false);
}

// reductions across the 16 lanes sharing a row-set (wave32, xor stays in half)
__device__ inline float red_max16(float v) {
    v = fmaxf(v, __shfl_xor(v, 1, 32));
    v = fmaxf(v, __shfl_xor(v, 2, 32));
    v = fmaxf(v, __shfl_xor(v, 4, 32));
    v = fmaxf(v, __shfl_xor(v, 8, 32));
    return v;
}
__device__ inline float red_sum16(float v) {
    v += __shfl_xor(v, 1, 32);
    v += __shfl_xor(v, 2, 32);
    v += __shfl_xor(v, 4, 32);
    v += __shfl_xor(v, 8, 32);
    return v;
}

// ---------------- fp32 -> bf16 convert ----------------
__global__ __launch_bounds__(256)
void cvt_kernel(const float* __restrict__ in, __bf16* __restrict__ out, int n) {
    int i = blockIdx.x * blockDim.x + threadIdx.x;
    if (i < n) out[i] = (__bf16)in[i];
}

// ============================================================
// Fused QKV projection.  Y = X @ W^T + b.
// X:[MTOT,EMB] bf16 row-major, W:[EMB,EMB] bf16 row-major [out,in].
// blockIdx.z selects {Q,K,V}. Q,K stored [b,h,n,d]; V stored [b,h,d,n].
// Wave tile: 64(M) x 64(N); 4 waves/block -> block tile 256 x 64.
// ============================================================
__global__ __launch_bounds__(128)
void qkv_gemm(const __bf16* __restrict__ X,
              const __bf16* __restrict__ Wq, const __bf16* __restrict__ Wk,
              const __bf16* __restrict__ Wv,
              const float* __restrict__ bq, const float* __restrict__ bk,
              const float* __restrict__ bv,
              __bf16* __restrict__ Qo, __bf16* __restrict__ Ko,
              __bf16* __restrict__ Vt) {
    const int lane = threadIdx.x & 31;
    const int wave = threadIdx.x >> 5;
    const int half = lane >> 4;
    const int lr   = lane & 15;
    const int mode = blockIdx.z;

    const __bf16* W; const float* bias; __bf16* dst;
    if (mode == 0)      { W = Wq; bias = bq; dst = Qo; }
    else if (mode == 1) { W = Wk; bias = bk; dst = Ko; }
    else                { W = Wv; bias = bv; dst = Vt; }

    const int m0 = blockIdx.y * 256 + wave * 64;
    const int n0 = blockIdx.x * 64;

    const __bf16* aRow[4];
#pragma unroll
    for (int mi = 0; mi < 4; mi++)
        aRow[mi] = X + (size_t)(m0 + mi * 16 + lr) * EMB;
    const __bf16* wCol[4];
#pragma unroll
    for (int t = 0; t < 4; t++)
        wCol[t] = W + (size_t)(n0 + t * 16 + lr) * EMB;

    v8f acc[4][4] = {};
    for (int k0 = 0; k0 < EMB; k0 += 32) {
        v16bf a[4];
#pragma unroll
        for (int mi = 0; mi < 4; mi++)
            a[mi] = load_fragA(aRow[mi] + k0, half);
        if (k0 + 32 < EMB) {
#pragma unroll
            for (int mi = 0; mi < 4; mi++)
                __builtin_prefetch(aRow[mi] + k0 + 32, 0, 3);
        }
#pragma unroll
        for (int t = 0; t < 4; t++) {
            v16bf b = load_fragB(wCol[t] + k0, half);
#pragma unroll
            for (int mi = 0; mi < 4; mi++)
                acc[mi][t] = wmma_bf16(a[mi], b, acc[mi][t]);
        }
    }

    const int bIdx = m0 >> 11;               // batch index (constant per wave)
#pragma unroll
    for (int t = 0; t < 4; t++) {
        const int n    = n0 + t * 16 + lr;
        const int head = n >> 6;
        const int d    = n & (HD - 1);
        const float bv_ = bias[n];
#pragma unroll
        for (int mi = 0; mi < 4; mi++) {
#pragma unroll
            for (int r = 0; r < 8; r++) {
                const int m  = m0 + mi * 16 + 8 * half + r;
                const int nn = m & (SEQ - 1);
                const float v = acc[mi][t][r] + bv_;
                size_t idx;
                if (mode == 2)
                    idx = (((size_t)(bIdx * NH + head)) * HD + d) * SEQ + nn;   // V^T
                else
                    idx = (((size_t)(bIdx * NH + head)) * SEQ + nn) * HD + d;   // Q,K
                dst[idx] = (__bf16)v;
            }
        }
    }
}

// ============================================================
// Flash attention core.
// One wave owns a 32-row q tile; 4 waves/block -> 128 q rows per block.
// grid: x = SEQ/128 = 16, y = B*H = 32.
// ============================================================
__global__ __launch_bounds__(128)
void attn_kernel(const __bf16* __restrict__ Q, const __bf16* __restrict__ Kt,
                 const __bf16* __restrict__ Vt, __bf16* __restrict__ O) {
    __shared__ __align__(16) __bf16 Pl[4][32 * 32];   // per-wave P staging (2 KB each)

    const int lane = threadIdx.x & 31;
    const int wave = threadIdx.x >> 5;
    const int half = lane >> 4;
    const int lr   = lane & 15;

    const int bh = blockIdx.y;
    const int q0 = blockIdx.x * 128 + wave * 32;

    const __bf16* Qh = Q  + (size_t)bh * SEQ * HD;
    const __bf16* Kh = Kt + (size_t)bh * SEQ * HD;
    const __bf16* Vh = Vt + (size_t)bh * HD * SEQ;

    // Q fragments: 2 M-subtiles x 2 K-steps (d = 0..31, 32..63)
    v16bf qf[2][2];
#pragma unroll
    for (int mi = 0; mi < 2; mi++) {
        qf[mi][0] = load_fragA(Qh + (size_t)(q0 + mi * 16 + lr) * HD + 0,  half);
        qf[mi][1] = load_fragA(Qh + (size_t)(q0 + mi * 16 + lr) * HD + 32, half);
    }

    float mrow[2][8], lrow[2][8];
#pragma unroll
    for (int mi = 0; mi < 2; mi++)
#pragma unroll
        for (int r = 0; r < 8; r++) { mrow[mi][r] = -3.0e38f; lrow[mi][r] = 0.0f; }
    v8f acc[2][4] = {};

    __bf16* pw = &Pl[wave][0];
    const float scale = 0.03125f;   // 1/sqrt(EMB) = 1/32

    for (int kv0 = 0; kv0 < SEQ; kv0 += 32) {
        if (kv0 + 32 < SEQ) {
            __builtin_prefetch(Kh + (size_t)(kv0 + 32 + lane) * HD, 0, 3);
            __builtin_prefetch(Vh + (size_t)(2 * lane) * SEQ + kv0 + 32, 0, 3);
        }
        // ---- S = Q K^T : 2 col-tiles x 2 M-subtiles, K=d in 2 steps ----
        v16bf b00 = load_fragB(Kh + (size_t)(kv0 + lr)      * HD + 0,  half);
        v16bf b01 = load_fragB(Kh + (size_t)(kv0 + lr)      * HD + 32, half);
        v16bf b10 = load_fragB(Kh + (size_t)(kv0 + 16 + lr) * HD + 0,  half);
        v16bf b11 = load_fragB(Kh + (size_t)(kv0 + 16 + lr) * HD + 32, half);
        v8f s[2][2];
#pragma unroll
        for (int mi = 0; mi < 2; mi++) {
            v8f z0 = {}, z1 = {};
            z0 = wmma_bf16(qf[mi][0], b00, z0);
            z0 = wmma_bf16(qf[mi][1], b01, z0);
            z1 = wmma_bf16(qf[mi][0], b10, z1);
            z1 = wmma_bf16(qf[mi][1], b11, z1);
            s[mi][0] = z0; s[mi][1] = z1;
        }

        // ---- online softmax (per subtile: rows = 8*half + r) ----
#pragma unroll
        for (int mi = 0; mi < 2; mi++) {
#pragma unroll
            for (int r = 0; r < 8; r++) {
                const float a0 = s[mi][0][r] * scale;
                const float a1 = s[mi][1][r] * scale;
                float mx = red_max16(fmaxf(a0, a1));
                const float mn = fmaxf(mrow[mi][r], mx);
                const float alpha = __expf(mrow[mi][r] - mn);
                const float e0 = __expf(a0 - mn);
                const float e1 = __expf(a1 - mn);
                const float rs = red_sum16(e0 + e1);
                lrow[mi][r] = lrow[mi][r] * alpha + rs;
                mrow[mi][r] = mn;
                acc[mi][0][r] *= alpha; acc[mi][1][r] *= alpha;
                acc[mi][2][r] *= alpha; acc[mi][3][r] *= alpha;
                // stash P back into s for the LDS restage
                s[mi][0][r] = e0; s[mi][1][r] = e1;
            }
        }

        // ---- restage P (D-layout -> A-layout) through per-wave LDS ----
#pragma unroll
        for (int mi = 0; mi < 2; mi++) {
#pragma unroll
            for (int r = 0; r < 8; r++) {
                const int row = mi * 16 + 8 * half + r;
                pw[row * 32 + lr]      = (__bf16)s[mi][0][r];
                pw[row * 32 + 16 + lr] = (__bf16)s[mi][1][r];
            }
        }
        asm volatile("s_wait_dscnt 0" ::: "memory");   // same-wave producer/consumer
        __builtin_amdgcn_wave_barrier();
        v16bf pa[2];
        pa[0] = load_fragA(pw + (0  + lr) * 32, half);
        pa[1] = load_fragA(pw + (16 + lr) * 32, half);
        __builtin_amdgcn_wave_barrier();

        // ---- O += P @ V  (B-frags contiguous thanks to V^T layout) ----
#pragma unroll
        for (int t = 0; t < 4; t++) {
            v16bf vb = load_fragB(Vh + (size_t)(t * 16 + lr) * SEQ + kv0, half);
#pragma unroll
            for (int mi = 0; mi < 2; mi++)
                acc[mi][t] = wmma_bf16(pa[mi], vb, acc[mi][t]);
        }
    }

    // ---- normalize and store merged-head O: [b, n, h*64 + d] bf16 ----
    const int b = bh >> 4;
    const int h = bh & (NH - 1);
#pragma unroll
    for (int mi = 0; mi < 2; mi++) {
#pragma unroll
        for (int r = 0; r < 8; r++) {
            const float inv = 1.0f / lrow[mi][r];
            const int q = q0 + mi * 16 + 8 * half + r;
            const size_t base = ((size_t)(b * SEQ + q)) * EMB + h * HD;
#pragma unroll
            for (int t = 0; t < 4; t++)
                O[base + t * 16 + lr] = (__bf16)(acc[mi][t][r] * inv);
        }
    }
}

// ============================================================
// Output projection: out = O @ Wo^T + bo (fp32 out).
// Wave tile 64 x 64; 4 waves/block -> block tile 256 x 64.
// ============================================================
__global__ __launch_bounds__(128)
void out_gemm(const __bf16* __restrict__ A, const __bf16* __restrict__ W,
              const float* __restrict__ bias, float* __restrict__ out) {
    const int lane = threadIdx.x & 31;
    const int wave = threadIdx.x >> 5;
    const int half = lane >> 4;
    const int lr   = lane & 15;

    const int m0 = blockIdx.y * 256 + wave * 64;
    const int n0 = blockIdx.x * 64;

    const __bf16* aRow[4];
#pragma unroll
    for (int mi = 0; mi < 4; mi++)
        aRow[mi] = A + (size_t)(m0 + mi * 16 + lr) * EMB;
    const __bf16* wCol[4];
#pragma unroll
    for (int t = 0; t < 4; t++)
        wCol[t] = W + (size_t)(n0 + t * 16 + lr) * EMB;

    v8f acc[4][4] = {};
    for (int k0 = 0; k0 < EMB; k0 += 32) {
        v16bf a[4];
#pragma unroll
        for (int mi = 0; mi < 4; mi++)
            a[mi] = load_fragA(aRow[mi] + k0, half);
        if (k0 + 32 < EMB) {
#pragma unroll
            for (int mi = 0; mi < 4; mi++)
                __builtin_prefetch(aRow[mi] + k0 + 32, 0, 3);
        }
#pragma unroll
        for (int t = 0; t < 4; t++) {
            v16bf b = load_fragB(wCol[t] + k0, half);
#pragma unroll
            for (int mi = 0; mi < 4; mi++)
                acc[mi][t] = wmma_bf16(a[mi], b, acc[mi][t]);
        }
    }
#pragma unroll
    for (int t = 0; t < 4; t++) {
        const int n = n0 + t * 16 + lr;
        const float bv_ = bias[n];
#pragma unroll
        for (int mi = 0; mi < 4; mi++) {
#pragma unroll
            for (int r = 0; r < 8; r++) {
                const int m = m0 + mi * 16 + 8 * half + r;
                out[(size_t)m * EMB + n] = acc[mi][t][r] + bv_;
            }
        }
    }
}

// ---------------- host launch ----------------
extern "C" void kernel_launch(void* const* d_in, const int* in_sizes, int n_in,
                              void* d_out, int out_size, void* d_ws, size_t ws_size,
                              hipStream_t stream) {
    (void)in_sizes; (void)n_in; (void)out_size; (void)ws_size;

    const float* x  = (const float*)d_in[0];
    const float* Wq = (const float*)d_in[1];
    const float* bq = (const float*)d_in[2];
    const float* Wk = (const float*)d_in[3];
    const float* bk = (const float*)d_in[4];
    const float* Wv = (const float*)d_in[5];
    const float* bv = (const float*)d_in[6];
    const float* Wo = (const float*)d_in[7];
    const float* bo = (const float*)d_in[8];

    char* ws = (char*)d_ws;
    __bf16* Xb  = (__bf16*)(ws);                         //  8 MiB: [4096,1024]
    __bf16* Wqb = (__bf16*)(ws + ((size_t)8  << 20));    //  2 MiB
    __bf16* Wkb = (__bf16*)(ws + ((size_t)10 << 20));
    __bf16* Wvb = (__bf16*)(ws + ((size_t)12 << 20));
    __bf16* Wob = (__bf16*)(ws + ((size_t)14 << 20));
    __bf16* Qb  = (__bf16*)(ws + ((size_t)16 << 20));    //  8 MiB: [b,h,n,d]
    __bf16* Kb  = (__bf16*)(ws + ((size_t)24 << 20));    //  8 MiB: [b,h,n,d]
    __bf16* Vt  = (__bf16*)(ws + ((size_t)32 << 20));    //  8 MiB: [b,h,d,n]
    __bf16* Ob  = (__bf16*)(ws + ((size_t)40 << 20));    //  8 MiB: [4096,1024]

    cvt_kernel<<<(MTOT * EMB) / 256, 256, 0, stream>>>(x,  Xb,  MTOT * EMB);
    cvt_kernel<<<(EMB * EMB)  / 256, 256, 0, stream>>>(Wq, Wqb, EMB * EMB);
    cvt_kernel<<<(EMB * EMB)  / 256, 256, 0, stream>>>(Wk, Wkb, EMB * EMB);
    cvt_kernel<<<(EMB * EMB)  / 256, 256, 0, stream>>>(Wv, Wvb, EMB * EMB);
    cvt_kernel<<<(EMB * EMB)  / 256, 256, 0, stream>>>(Wo, Wob, EMB * EMB);

    dim3 g1(EMB / 64, MTOT / 256, 3);
    qkv_gemm<<<g1, 128, 0, stream>>>(Xb, Wqb, Wkb, Wvb, bq, bk, bv, Qb, Kb, Vt);

    dim3 g2(SEQ / 128, BB * NH);
    attn_kernel<<<g2, 128, 0, stream>>>(Qb, Kb, Vt, Ob);

    dim3 g3(EMB / 64, MTOT / 256);
    out_gemm<<<g3, 128, 0, stream>>>(Ob, Wob, bo, (float*)d_out);
}
